// GrpahWeightRegressor_res_gat_35716948033584
// MI455X (gfx1250) — matrix operand order repaired and checked by reading, not compiled, not run on hardware
//
#include <hip/hip_runtime.h>
#include <hip/hip_bf16.h>

// ---------------------------------------------------------------------------
// 7-layer DeepGCN/GAT. N=20000, D=512, H=8 x C=64, 180k edges (w/ self loops).
//   - dense GEMM: v_wmma_f32_16x16x32_bf16, K double-buffered through LDS via
//     GLOBAL_LOAD_ASYNC_TO_LDS_B128 (ASYNCcnt) when available
//   - edge softmax: float atomic-max (int trick) + atomic adds
//   - aggregation: f32 global atomics (HBM-bound phase, ~6-7 GB total)
//   - BN column stats via partial-sum atomics; fused BN+ReLU+GEMV finale
// ---------------------------------------------------------------------------

#define NN      20000
#define EE      160000
#define ETOT    180000   // EE + NN self loops
#define DD      512
#define HH      8
#define CC      64
#define LL      7
#define BNEPS   1e-5f
#define ASTRIDE 40       // LDS row stride (shorts): 80B = 16B-aligned, no bank conflicts

#if defined(__gfx1250__) && __has_builtin(__builtin_amdgcn_global_load_async_to_lds_b128) && __has_builtin(__builtin_amdgcn_s_wait_asynccnt)
#define USE_ASYNC_LDS 1
typedef int i32x4 __attribute__((vector_size(16)));
typedef __attribute__((address_space(1))) i32x4 glob_i32x4;
typedef __attribute__((address_space(3))) i32x4 lds_i32x4;
#else
#define USE_ASYNC_LDS 0
#endif

typedef __attribute__((ext_vector_type(16))) __bf16 bf16x16;
typedef __attribute__((ext_vector_type(8)))  float  f32x8;

__device__ __forceinline__ unsigned short f32_to_bf16(float f) {
  unsigned int x = __float_as_uint(f);
  x += 0x7FFFu + ((x >> 16) & 1u);          // round-to-nearest-even
  return (unsigned short)(x >> 16);
}

__device__ __forceinline__ void atomicMaxF32(float* addr, float v) {
  if (v >= 0.0f) atomicMax((int*)addr, __float_as_int(v));
  else           atomicMin((unsigned int*)addr, __float_as_uint(v));
}

// ------------------------------- casts -------------------------------------
__global__ void cast_f32_bf16_kernel(const float* __restrict__ in,
                                     unsigned short* __restrict__ out, int n) {
  int i = blockIdx.x * blockDim.x + threadIdx.x;
  if (i < n) out[i] = f32_to_bf16(in[i]);
}

// Wt[l][n][k] = bf16(Ws[l][k][n])  (transposed weights -> DMA-friendly B tiles)
__global__ void transpose_cast_w_kernel(const float* __restrict__ Ws,
                                        unsigned short* __restrict__ Wt) {
  int i = blockIdx.x * blockDim.x + threadIdx.x;
  if (i >= LL * DD * DD) return;
  int l = i / (DD * DD), r = i % (DD * DD);
  int n = r / DD, k = r % DD;
  Wt[i] = f32_to_bf16(Ws[l * DD * DD + k * DD + n]);
}

// ------------------------------- GEMM --------------------------------------
// C[NN x 512] = A(bf16)[NN x 512] * W[512 x 512]; Wt passed transposed [n][k].
// 256 threads = 8 waves (2Mx4N); block tile 64x128; wave tile 32x32 via 2x2
// v_wmma_f32_16x16x32_bf16. K staged 32 wide, double-buffered in LDS with
// async global->LDS B128 copies overlapping WMMA of the previous slab.
__global__ void __launch_bounds__(256)
gemm_bf16_kernel(const unsigned short* __restrict__ A,
                 const unsigned short* __restrict__ Wt,
                 float* __restrict__ Cout) {
  __shared__ __align__(16) unsigned short As[2][64 * ASTRIDE];
  __shared__ __align__(16) unsigned short Bs[2][128 * ASTRIDE];

  const int t      = threadIdx.x;
  const int lane   = t & 31;
  const int waveId = t >> 5;
  const int wr     = waveId >> 2;       // 0..1  (M)
  const int wc     = waveId & 3;        // 0..3  (N)
  const int hf     = lane >> 4;
  const int lanelo = lane & 15;

  const int rowBase = blockIdx.x * 64;
  const int colBase = blockIdx.y * 128;

  // per-thread 16B copy slots
  const int arow = t >> 2, ach = t & 3;                 // A: 64 rows x 4 chunks
  const bool aok = (rowBase + arow) < NN;
  const unsigned short* agp = A + (rowBase + arow) * DD + ach * 8;
  unsigned short* alp = &As[0][0] + arow * ASTRIDE + ach * 8;

  const int bn0 = t >> 2, bch0 = t & 3;                 // B: 128 rows x 4 chunks,
  const int bn1 = (t + 256) >> 2, bch1 = (t + 256) & 3; // 2 slots per thread
  const unsigned short* bgp0 = Wt + (colBase + bn0) * DD + bch0 * 8;
  const unsigned short* bgp1 = Wt + (colBase + bn1) * DD + bch1 * 8;
  unsigned short* blp0 = &Bs[0][0] + bn0 * ASTRIDE + bch0 * 8;
  unsigned short* blp1 = &Bs[0][0] + bn1 * ASTRIDE + bch1 * 8;

  const int bufStrideA = 64 * ASTRIDE;
  const int bufStrideB = 128 * ASTRIDE;

  auto issueTile = [&](int kt, int buf) {
    const int k0 = kt * 32;
#if USE_ASYNC_LDS
    if (aok)
      __builtin_amdgcn_global_load_async_to_lds_b128(
          (glob_i32x4*)(void*)(agp + k0),
          (lds_i32x4*)(void*)(alp + buf * bufStrideA), 0, 0);
    __builtin_amdgcn_global_load_async_to_lds_b128(
        (glob_i32x4*)(void*)(bgp0 + k0),
        (lds_i32x4*)(void*)(blp0 + buf * bufStrideB), 0, 0);
    __builtin_amdgcn_global_load_async_to_lds_b128(
        (glob_i32x4*)(void*)(bgp1 + k0),
        (lds_i32x4*)(void*)(blp1 + buf * bufStrideB), 0, 0);
#else
    if (aok) *(uint4*)(alp + buf * bufStrideA) = *(const uint4*)(agp + k0);
    *(uint4*)(blp0 + buf * bufStrideB) = *(const uint4*)(bgp0 + k0);
    *(uint4*)(blp1 + buf * bufStrideB) = *(const uint4*)(bgp1 + k0);
#endif
  };

  f32x8 acc[2][2];
#pragma unroll
  for (int mi = 0; mi < 2; ++mi)
#pragma unroll
    for (int ni = 0; ni < 2; ++ni) acc[mi][ni] = (f32x8)0.0f;

  issueTile(0, 0);

  for (int kt = 0; kt < 16; ++kt) {
    const int buf = kt & 1;
#if USE_ASYNC_LDS
    __builtin_amdgcn_s_wait_asynccnt(0);
#endif
    __syncthreads();
    if (kt + 1 < 16) issueTile(kt + 1, buf ^ 1);

    // fragments per documented 16-bit WMMA layouts
    union Frag { bf16x16 v; unsigned int u[8]; };
    Frag af[2], bf[2];
#pragma unroll
    for (int mi = 0; mi < 2; ++mi) {
      const unsigned short* ap = &As[buf][(wr * 32 + mi * 16 + lanelo) * ASTRIDE];
#pragma unroll
      for (int v = 0; v < 8; ++v) {
        int grp = v >> 2, vv = v & 3;        // K = grp*16 + hf*8 + vv*2 (+sub)
        af[mi].u[v] = *(const unsigned int*)(ap + grp * 16 + hf * 8 + vv * 2);
      }
    }
#pragma unroll
    for (int ni = 0; ni < 2; ++ni) {
      const unsigned short* bp = &Bs[buf][(wc * 32 + ni * 16 + lanelo) * ASTRIDE];
#pragma unroll
      for (int v = 0; v < 8; ++v)            // K = (v>>2)*16 + hf*8 + (v&3)*2 (+sub)
        bf[ni].u[v] = *(const unsigned int*)(bp + (v >> 2) * 16 + hf * 8 + (v & 3) * 2);
    }
#pragma unroll
    for (int mi = 0; mi < 2; ++mi)
#pragma unroll
      for (int ni = 0; ni < 2; ++ni)
        acc[mi][ni] = __builtin_amdgcn_wmma_f32_16x16x32_bf16(
            false, af[mi].v, false, bf[ni].v, (short)0, acc[mi][ni],
            false, false);
  }

  // store: C/D layout: VGPR r -> M = r + 8*half, N = lanelo
#pragma unroll
  for (int mi = 0; mi < 2; ++mi)
#pragma unroll
    for (int ni = 0; ni < 2; ++ni)
#pragma unroll
      for (int r = 0; r < 8; ++r) {
        int row = rowBase + wr * 32 + mi * 16 + hf * 8 + r;
        int col = colBase + wc * 32 + ni * 16 + lanelo;
        if (row < NN) Cout[row * DD + col] = acc[mi][ni][r];
      }
}

// ------------------------------ attention ----------------------------------
__global__ void scores_kernel(const float* __restrict__ hW,
                              const float* __restrict__ asrc,
                              const float* __restrict__ adst,
                              float* __restrict__ ssrc,
                              float* __restrict__ sdst) {
  int idx = blockIdx.x * blockDim.x + threadIdx.x;
  if (idx >= NN * HH) return;
  int n = idx >> 3, hh = idx & 7;
  const float4* hp = (const float4*)(hW + n * DD + hh * CC);
  const float4* ap = (const float4*)(asrc + hh * CC);
  const float4* bp = (const float4*)(adst + hh * CC);
  float s1 = 0.f, s2 = 0.f;
#pragma unroll
  for (int i = 0; i < 16; ++i) {
    float4 hv = hp[i], av = ap[i], bv = bp[i];
    s1 += hv.x * av.x + hv.y * av.y + hv.z * av.z + hv.w * av.w;
    s2 += hv.x * bv.x + hv.y * bv.y + hv.z * bv.z + hv.w * bv.w;
  }
  ssrc[idx] = s1;
  sdst[idx] = s2;
}

__global__ void init_softmax_kernel(float* __restrict__ amax,
                                    float* __restrict__ den) {
  int i = blockIdx.x * blockDim.x + threadIdx.x;
  if (i < NN * HH) { amax[i] = -3.0e38f; den[i] = 0.f; }
}

__global__ void edge_alpha_kernel(const int* __restrict__ ei,
                                  const float* __restrict__ ssrc,
                                  const float* __restrict__ sdst,
                                  float* __restrict__ alpha,
                                  float* __restrict__ amax) {
  int idx = blockIdx.x * blockDim.x + threadIdx.x;
  if (idx >= ETOT * HH) return;
  int e = idx >> 3, hh = idx & 7;
  int s, d;
  if (e < EE) { s = ei[e]; d = ei[EE + e]; } else { s = d = e - EE; }
  float a = ssrc[s * HH + hh] + sdst[d * HH + hh];
  a = (a >= 0.f) ? a : 0.2f * a;                 // leaky_relu(0.2)
  alpha[idx] = a;
  atomicMaxF32(&amax[d * HH + hh], a);
}

__global__ void edge_exp_kernel(const int* __restrict__ ei,
                                const float* __restrict__ alpha,
                                const float* __restrict__ amax,
                                float* __restrict__ exb,
                                float* __restrict__ den) {
  int idx = blockIdx.x * blockDim.x + threadIdx.x;
  if (idx >= ETOT * HH) return;
  int e = idx >> 3, hh = idx & 7;
  int d = (e < EE) ? ei[EE + e] : (e - EE);
  float ex = __expf(alpha[idx] - amax[d * HH + hh]);
  exb[idx] = ex;
  atomicAdd(&den[d * HH + hh], ex);
}

__global__ void init_acc_kernel(const float* __restrict__ resid,
                                const float* __restrict__ bias,
                                float* __restrict__ acc) {
  int i = blockIdx.x * blockDim.x + threadIdx.x;
  if (i >= NN * DD) return;
  float v = bias[i & (DD - 1)];
  if (resid) v += resid[i];
  acc[i] = v;
}

__global__ void aggregate_kernel(const int* __restrict__ ei,
                                 const float* __restrict__ hW,
                                 const float* __restrict__ exb,
                                 const float* __restrict__ den,
                                 float* __restrict__ acc) {
  int idx = blockIdx.x * blockDim.x + threadIdx.x;
  if (idx >= ETOT * DD) return;
  int e = idx >> 9, j = idx & (DD - 1);
  int hh = j >> 6;
  int s, d;
  if (e < EE) { s = ei[e]; d = ei[EE + e]; } else { s = d = e - EE; }
  float w = exb[e * HH + hh] / (den[d * HH + hh] + 1e-16f);
  atomicAdd(&acc[d * DD + j], hW[s * DD + j] * w);
}

// ------------------------------ batch norm ---------------------------------
__global__ void zero512_kernel(float* __restrict__ a, float* __restrict__ b) {
  int j = threadIdx.x;
  a[j] = 0.f; b[j] = 0.f;
}

__global__ void bn_stats_kernel(const float* __restrict__ h,
                                float* __restrict__ colsum,
                                float* __restrict__ colsq) {
  int j  = threadIdx.x;           // 512 threads = columns
  int r0 = blockIdx.x * 100;      // 200 blocks x 100 rows
  float s = 0.f, s2 = 0.f;
  for (int r = r0; r < r0 + 100; ++r) {
    float v = h[r * DD + j];
    s += v; s2 += v * v;
  }
  atomicAdd(&colsum[j], s);
  atomicAdd(&colsq[j], s2);
}

__global__ void bn_finalize_kernel(const float* __restrict__ colsum,
                                   const float* __restrict__ colsq,
                                   float* __restrict__ mean,
                                   float* __restrict__ rstd) {
  int j = threadIdx.x;
  float mu  = colsum[j] * (1.0f / NN);
  float var = colsq[j] * (1.0f / NN) - mu * mu;
  mean[j] = mu;
  rstd[j] = rsqrtf(var + BNEPS);
}

__global__ void bn_relu_cast_kernel(const float* __restrict__ h,
                                    const float* __restrict__ g,
                                    const float* __restrict__ b,
                                    const float* __restrict__ mean,
                                    const float* __restrict__ rstd,
                                    unsigned short* __restrict__ zb) {
  int i = blockIdx.x * blockDim.x + threadIdx.x;
  if (i >= NN * DD) return;
  int j = i & (DD - 1);
  float z = g[j] * (h[i] - mean[j]) * rstd[j] + b[j];
  z = (z > 0.f) ? z : 0.f;
  zb[i] = f32_to_bf16(z);
}

// ------------------------------ final GEMV ---------------------------------
__global__ void final_kernel(const float* __restrict__ h,
                             const float* __restrict__ g,
                             const float* __restrict__ b,
                             const float* __restrict__ mean,
                             const float* __restrict__ rstd,
                             const float* __restrict__ lw,
                             const float* __restrict__ lb,
                             float* __restrict__ out) {
  __shared__ float red[256];
  int n = blockIdx.x, t = threadIdx.x;
  float s = 0.f;
  for (int j = t; j < DD; j += 256) {
    float z = g[j] * (h[n * DD + j] - mean[j]) * rstd[j] + b[j];
    z = (z > 0.f) ? z : 0.f;
    s += z * lw[j];
  }
  red[t] = s;
  __syncthreads();
  for (int st = 128; st > 0; st >>= 1) {
    if (t < st) red[t] += red[t + st];
    __syncthreads();
  }
  if (t == 0) out[n] = red[0] + lb[0];
}

// ------------------------------ launcher -----------------------------------
extern "C" void kernel_launch(void* const* d_in, const int* in_sizes, int n_in,
                              void* d_out, int out_size, void* d_ws, size_t ws_size,
                              hipStream_t stream) {
  const float* x        = (const float*)d_in[0];
  const int*   ei       = (const int*)d_in[1];
  const float* Ws       = (const float*)d_in[2];
  const float* att_src  = (const float*)d_in[3];
  const float* att_dst  = (const float*)d_in[4];
  const float* conv_b   = (const float*)d_in[5];
  const float* bn_g     = (const float*)d_in[6];
  const float* bn_b     = (const float*)d_in[7];
  const float* lin_w    = (const float*)d_in[8];
  const float* lin_b    = (const float*)d_in[9];
  float* out = (float*)d_out;
  (void)in_sizes; (void)n_in; (void)out_size; (void)ws_size;

  char* base = (char*)d_ws;
  size_t off = 0;
  auto carve = [&](size_t bytes) -> void* {
    void* p = base + off;
    off += (bytes + 255) & ~(size_t)255;
    return p;
  };
  unsigned short* Wt   = (unsigned short*)carve((size_t)LL * DD * DD * 2);
  unsigned short* zb   = (unsigned short*)carve((size_t)NN * DD * 2);
  float* hW    = (float*)carve((size_t)NN * DD * 4);
  float* hA    = (float*)carve((size_t)NN * DD * 4);
  float* hB    = (float*)carve((size_t)NN * DD * 4);
  float* ssrc  = (float*)carve((size_t)NN * HH * 4);
  float* sdst  = (float*)carve((size_t)NN * HH * 4);
  float* amax  = (float*)carve((size_t)NN * HH * 4);
  float* den   = (float*)carve((size_t)NN * HH * 4);
  float* alpha = (float*)carve((size_t)ETOT * HH * 4);
  float* exb   = (float*)carve((size_t)ETOT * HH * 4);
  float* colsum = (float*)carve(DD * 4);
  float* colsq  = (float*)carve(DD * 4);
  float* mean   = (float*)carve(DD * 4);
  float* rstd   = (float*)carve(DD * 4);

  // transposed bf16 weights, once for all layers
  transpose_cast_w_kernel<<<(LL * DD * DD + 255) / 256, 256, 0, stream>>>(Ws, Wt);

  dim3 ggrid((NN + 63) / 64, DD / 128);

  auto conv = [&](const unsigned short* zin, int l, const float* resid,
                  float* acc) {
    gemm_bf16_kernel<<<ggrid, 256, 0, stream>>>(zin, Wt + (size_t)l * DD * DD, hW);
    scores_kernel<<<(NN * HH + 255) / 256, 256, 0, stream>>>(
        hW, att_src + l * HH * CC, att_dst + l * HH * CC, ssrc, sdst);
    init_softmax_kernel<<<(NN * HH + 255) / 256, 256, 0, stream>>>(amax, den);
    edge_alpha_kernel<<<(ETOT * HH + 255) / 256, 256, 0, stream>>>(
        ei, ssrc, sdst, alpha, amax);
    edge_exp_kernel<<<(ETOT * HH + 255) / 256, 256, 0, stream>>>(
        ei, alpha, amax, exb, den);
    init_acc_kernel<<<(NN * DD + 255) / 256, 256, 0, stream>>>(
        resid, conv_b + l * DD, acc);
    aggregate_kernel<<<(ETOT * DD + 255) / 256, 256, 0, stream>>>(
        ei, hW, exb, den, acc);
  };

  // layer 0: h = conv(x)
  cast_f32_bf16_kernel<<<(NN * DD + 255) / 256, 256, 0, stream>>>(x, zb, NN * DD);
  conv(zb, 0, nullptr, hA);
  float* h = hA;
  float* nxt = hB;

  // layers 1..6: h = h + conv(relu(bn(h)))
  for (int l = 1; l < LL; ++l) {
    zero512_kernel<<<1, 512, 0, stream>>>(colsum, colsq);
    bn_stats_kernel<<<200, 512, 0, stream>>>(h, colsum, colsq);
    bn_finalize_kernel<<<1, 512, 0, stream>>>(colsum, colsq, mean, rstd);
    bn_relu_cast_kernel<<<(NN * DD + 255) / 256, 256, 0, stream>>>(
        h, bn_g + l * DD, bn_b + l * DD, mean, rstd, zb);
    conv(zb, l, h, nxt);
    float* tmp = h; h = nxt; nxt = tmp;
  }

  // final: relu(bn(h, gamma0, beta0)) @ lin_w + lin_b
  zero512_kernel<<<1, 512, 0, stream>>>(colsum, colsq);
  bn_stats_kernel<<<200, 512, 0, stream>>>(h, colsum, colsq);
  bn_finalize_kernel<<<1, 512, 0, stream>>>(colsum, colsq, mean, rstd);
  final_kernel<<<NN, 256, 0, stream>>>(h, bn_g, bn_b, mean, rstd, lin_w, lin_b, out);
}
